// DecoderRNN_36996848288017
// MI455X (gfx1250) — matrix operand order repaired and checked by reading, not compiled, not run on hardware
//
#include <hip/hip_runtime.h>
#include <math.h>
#include <stdint.h>

// ---------------------------------------------------------------------------
// Types for CDNA5 WMMA (wave32): v_wmma_f32_16x16x32_bf16
// ---------------------------------------------------------------------------
typedef __bf16 bf16;
typedef bf16  v8bf  __attribute__((ext_vector_type(8)));
typedef bf16  v16bf __attribute__((ext_vector_type(16)));
typedef float v8f   __attribute__((ext_vector_type(8)));

#define WMMA_BF16(A, B, C) \
    __builtin_amdgcn_wmma_f32_16x16x32_bf16(false, (A), false, (B), (short)0, (C), false, false)

// Problem dims
constexpr int kBatch = 512;
constexpr int kTenc  = 140;
constexpr int kE     = 256;
constexpr int kH     = 512;
constexpr int kO     = 64;
constexpr int kPred  = 140;
constexpr int kTfull = kTenc + kPred; // 280

// LDS row strides (padded by 8 bf16 = 16B so 16 rows hit distinct bank groups)
constexpr int XS_LD = kE + 8;  // 264
constexpr int HS_LD = kH + 8;  // 520
constexpr int OS_LD = kE + 8;  // 264

// ---------------------------------------------------------------------------
// CDNA5 async global->LDS copy (ASYNCcnt-tracked, bypasses VGPRs).
// Per-lane: copies 16 bytes from global address vaddr to LDS offset vdst. (CDNA5 ISA)
// ---------------------------------------------------------------------------
__device__ __forceinline__ void async_load_b128(void* lds_ptr, const void* gptr) {
    uint32_t lds_off = (uint32_t)(uintptr_t)lds_ptr;        // low 32b of generic = LDS offset
    uint64_t gaddr   = (uint64_t)(uintptr_t)gptr;
    asm volatile("global_load_async_to_lds_b128 %0, %1, off"
                 :: "v"(lds_off), "v"(gaddr) : "memory");
}
__device__ __forceinline__ void wait_asynccnt_0() {
    asm volatile("s_wait_asynccnt 0x0" ::: "memory");
}
__device__ __forceinline__ void wait_asynccnt_4() {
    asm volatile("s_wait_asynccnt 0x4" ::: "memory");
}

// ---------------------------------------------------------------------------
// A-fragment loader (16x32 bf16, contiguous k order, matching glb_b_frag):
//   lanes 0-15 : row = lane,    K = k0+{0..15}
//   lanes 16-31: row = lane-16, K = k0+{16..31}
// ---------------------------------------------------------------------------
__device__ __forceinline__ v16bf lds_a_frag(const bf16* rowPtr, int k0, int lane) {
    const int lh = (lane & 16) ? 16 : 0;
    v8bf lo = *(const v8bf*)(rowPtr + k0 + lh);
    v8bf hi = *(const v8bf*)(rowPtr + k0 + 8 + lh);
    return __builtin_shufflevector(lo, hi, 0,1,2,3,4,5,6,7,8,9,10,11,12,13,14,15);
}

// B-fragment (32x16 bf16): B(k,n) = W[n,k] with W row-major, so lane n%16 reads
// 16 contiguous bf16 of weight row n at K offset (lane<16 ? 0 : 16). One 32B load.
__device__ __forceinline__ v16bf glb_b_frag(const bf16* Wrow, int k0) {
    return *(const v16bf*)(Wrow + k0);
}

__device__ __forceinline__ float sigmoidf_(float x) {
    return 1.0f / (1.0f + __expf(-x));
}

// ---------------------------------------------------------------------------
// fp32 -> bf16 weight conversion (runs every launch; weights then live in L2)
// ---------------------------------------------------------------------------
__global__ void f32_to_bf16_kernel(const float* __restrict__ src, bf16* __restrict__ dst, int n) {
    int i = blockIdx.x * blockDim.x + threadIdx.x;
    if (i < n) dst[i] = (bf16)src[i];
}

// ---------------------------------------------------------------------------
// State init: x0 = encoder_outputs[:, -1, :], h0 = encoder_hidden[0]
// ---------------------------------------------------------------------------
__global__ void init_state_kernel(const float* __restrict__ enc, const float* __restrict__ h0,
                                  bf16* __restrict__ x_bf, float* __restrict__ h_f32,
                                  bf16* __restrict__ h_bf) {
    int i = blockIdx.x * blockDim.x + threadIdx.x;
    if (i < kBatch * kE) {
        int b = i / kE, e = i % kE;
        x_bf[i] = (bf16)enc[(size_t)b * (kTenc * kE) + (size_t)(kTenc - 1) * kE + e];
    }
    if (i < kBatch * kH) {
        float v = h0[i];
        h_f32[i] = v;
        h_bf[i]  = (bf16)v;
    }
}

// ---------------------------------------------------------------------------
// Encoder half of the regression: out[b, 0:140, :] = enc @ regW^T + reg_b
// 8 waves/WG, each wave owns one 16-row tile of the flattened (B*T_ENC) rows.
// ---------------------------------------------------------------------------
__global__ __launch_bounds__(256) void encoder_reg_kernel(
    const float* __restrict__ enc, const bf16* __restrict__ regWb,
    const float* __restrict__ reg_b, float* __restrict__ out_full) {
    const int tid  = threadIdx.x;
    const int lane = tid & 31;
    const int wave = tid >> 5;
    const int tile = blockIdx.x * 8 + wave;          // 16-row tile of 71680 rows
    const int rowBase = tile * 16;
    const int col  = lane & 15;
    const int lh   = (lane & 16) ? 16 : 0;
    const int rOff = (lane & 16) ? 8 : 0;

    const float* Arow = enc + (size_t)(rowBase + col) * kE;

    // Hoist A fragments for all 8 K-steps (K = 256)
    v16bf afr[8];
#pragma unroll
    for (int ks = 0; ks < 8; ++ks) {
        const int k0 = ks * 32;
        v16bf a;
#pragma unroll
        for (int j = 0; j < 8; ++j) a[j]     = (bf16)Arow[k0 + lh + j];
#pragma unroll
        for (int j = 0; j < 8; ++j) a[8 + j] = (bf16)Arow[k0 + 8 + lh + j];
        afr[ks] = a;
    }

#pragma unroll
    for (int nb = 0; nb < 4; ++nb) {                  // N = 64 -> 4 tiles
        const int n = nb * 16 + col;
        const bf16* Bp = regWb + (size_t)n * kE + ((lane & 16) ? 16 : 0);
        v8f acc = {};
#pragma unroll
        for (int ks = 0; ks < 8; ++ks)
            acc = WMMA_BF16(afr[ks], glb_b_frag(Bp, ks * 32), acc);
        const float bias = reg_b[n];
#pragma unroll
        for (int v = 0; v < 8; ++v) {
            const int rF = rowBase + v + rOff;        // flat row = b*140 + t
            const int b  = rF / kTenc;
            const int tt = rF % kTenc;
            out_full[(size_t)b * (kTfull * kO) + (size_t)tt * kO + n] = acc[v] + bias;
        }
    }
}

// ---------------------------------------------------------------------------
// One fused GRU step for a 16-row batch tile (32 WGs x 8 waves):
//   stage x (async, waited before A) and h (async, overlapped with A)
//   A: e = relu(x @ embW^T + emb_b)                       (LDS)
//   B: gates (r,z fused accumulators; gi_n/gh_n split) -> h_new, pipelined B-frags
//   C: out = h_new @ outW^T + out_b -> next x (bf16, global) + LDS
//   D: d_out[:, 140+t, :] = out @ regW^T + reg_b          (dec regression fused)
// ---------------------------------------------------------------------------
__global__ __launch_bounds__(256) void gru_step_kernel(
    const bf16* __restrict__ embWb, const bf16* __restrict__ wihb,
    const bf16* __restrict__ whhb,  const bf16* __restrict__ outWb,
    const bf16* __restrict__ regWb,
    const float* __restrict__ emb_b, const float* __restrict__ b_ih,
    const float* __restrict__ b_hh,  const float* __restrict__ out_b,
    const float* __restrict__ reg_b,
    bf16* __restrict__ x_bf, float* __restrict__ h_f32, bf16* __restrict__ h_bf,
    float* __restrict__ out_full, int t) {

    __shared__ bf16 xs [16 * XS_LD];   // x tile      16x256
    __shared__ bf16 es [16 * HS_LD];   // e tile      16x512
    __shared__ bf16 hs [16 * HS_LD];   // h tile      16x512
    __shared__ bf16 hns[16 * HS_LD];   // h_new tile  16x512
    __shared__ bf16 os [16 * OS_LD];   // out tile    16x256

    const int tid  = threadIdx.x;
    const int lane = tid & 31;
    const int wave = tid >> 5;
    const int rowBase = blockIdx.x * 16;
    const int col  = lane & 15;                       // N (and A-row) selector
    const int rOff = (lane & 16) ? 8 : 0;             // C/D row offset
    const int kOff = (lane & 16) ? 16 : 0;            // B K-offset

    // ---- async stage x then h into LDS (16B chunks, per-lane addresses) -----
    // x: 512 chunks -> 2 per thread (2 async instrs/wave)
#pragma unroll
    for (int j = 0; j < 2; ++j) {
        const int c = tid + j * 256;                  // chunk id
        const int r = c / (kE / 8), cc = (c % (kE / 8)) * 8;
        async_load_b128(&xs[r * XS_LD + cc], &x_bf[(size_t)(rowBase + r) * kE + cc]);
    }
    // h: 1024 chunks -> 4 per thread (4 async instrs/wave, issued after x)
#pragma unroll
    for (int j = 0; j < 4; ++j) {
        const int c = tid + j * 256;
        const int r = c / (kH / 8), cc = (c % (kH / 8)) * 8;
        async_load_b128(&hs[r * HS_LD + cc], &h_bf[(size_t)(rowBase + r) * kH + cc]);
    }
    wait_asynccnt_4();        // x chunks (2 oldest per wave) done; h still in flight
    __syncthreads();

    // ---- Phase A: e = relu(x @ embW^T + emb_b), N = 512 ---------------------
    // (h staging overlaps with this entire phase)
    for (int nb = wave; nb < kH / 16; nb += 8) {
        const int n = nb * 16 + col;
        const bf16* Bp = embWb + (size_t)n * kE + kOff;
        v8f acc = {};
#pragma unroll
        for (int ks = 0; ks < kE / 32; ++ks)
            acc = WMMA_BF16(lds_a_frag(&xs[col * XS_LD], ks * 32, lane),
                            glb_b_frag(Bp, ks * 32), acc);
        const float bias = emb_b[n];
#pragma unroll
        for (int v = 0; v < 8; ++v) {
            float val = acc[v] + bias;
            val = val > 0.0f ? val : 0.0f;
            es[(v + rOff) * HS_LD + n] = (bf16)val;
        }
    }
    wait_asynccnt_0();        // drain h staging before anyone reads hs
    __syncthreads();

    // ---- Phase B: gates + h_new, N = 512, B-frags software-pipelined --------
    for (int nb = wave; nb < kH / 16; nb += 8) {
        const int n = nb * 16 + col;
        const bf16* BiR = wihb + (size_t)(0 * kH + n) * kH + kOff;
        const bf16* BiZ = wihb + (size_t)(1 * kH + n) * kH + kOff;
        const bf16* BiN = wihb + (size_t)(2 * kH + n) * kH + kOff;
        const bf16* BhR = whhb + (size_t)(0 * kH + n) * kH + kOff;
        const bf16* BhZ = whhb + (size_t)(1 * kH + n) * kH + kOff;
        const bf16* BhN = whhb + (size_t)(2 * kH + n) * kH + kOff;
        v8f accR = {}, accZ = {}, accGN = {}, accHN = {};

        // prime the pipeline
        v16bf fiR = glb_b_frag(BiR, 0), fhR = glb_b_frag(BhR, 0);
        v16bf fiZ = glb_b_frag(BiZ, 0), fhZ = glb_b_frag(BhZ, 0);
        v16bf fiN = glb_b_frag(BiN, 0), fhN = glb_b_frag(BhN, 0);
#pragma unroll
        for (int ks = 0; ks < kH / 32; ++ks) {
            const int k0 = ks * 32;
            v16bf niR = fiR, nhR = fhR, niZ = fiZ, nhZ = fhZ, niN = fiN, nhN = fhN;
            if (ks + 1 < kH / 32) {                   // prefetch next K-step's B-frags
                const int kn = k0 + 32;
                niR = glb_b_frag(BiR, kn); nhR = glb_b_frag(BhR, kn);
                niZ = glb_b_frag(BiZ, kn); nhZ = glb_b_frag(BhZ, kn);
                niN = glb_b_frag(BiN, kn); nhN = glb_b_frag(BhN, kn);
            }
            v16bf ae = lds_a_frag(&es[col * HS_LD], k0, lane);
            v16bf ah = lds_a_frag(&hs[col * HS_LD], k0, lane);
            accR  = WMMA_BF16(ae, fiR, accR);         // r: gi+gh fused accumulator
            accR  = WMMA_BF16(ah, fhR, accR);
            accZ  = WMMA_BF16(ae, fiZ, accZ);         // z: gi+gh fused accumulator
            accZ  = WMMA_BF16(ah, fhZ, accZ);
            accGN = WMMA_BF16(ae, fiN, accGN);        // n: keep gi/gh split (r*hn)
            accHN = WMMA_BF16(ah, fhN, accHN);
            fiR = niR; fhR = nhR; fiZ = niZ; fhZ = nhZ; fiN = niN; fhN = nhN;
        }
        const float biR = b_ih[n]          + b_hh[n];
        const float biZ = b_ih[kH + n]     + b_hh[kH + n];
        const float biN = b_ih[2 * kH + n];
        const float bhN = b_hh[2 * kH + n];
#pragma unroll
        for (int v = 0; v < 8; ++v) {
            const int r = v + rOff;
            const float rg = sigmoidf_(accR[v] + biR);
            const float zg = sigmoidf_(accZ[v] + biZ);
            const float ng = tanhf(accGN[v] + biN + rg * (accHN[v] + bhN));
            const size_t gidx = (size_t)(rowBase + r) * kH + n;
            const float hold = h_f32[gidx];
            const float hnew = (1.0f - zg) * ng + zg * hold;
            h_f32[gidx] = hnew;                       // each element owned by one lane
            const bf16 hb = (bf16)hnew;
            h_bf[gidx] = hb;
            hns[r * HS_LD + n] = hb;
        }
    }
    __syncthreads();

    // ---- Phase C: out = h_new @ outW^T + out_b, N = 256 ---------------------
    for (int nb = wave; nb < kE / 16; nb += 8) {
        const int n = nb * 16 + col;
        const bf16* Bp = outWb + (size_t)n * kH + kOff;
        v8f acc = {};
#pragma unroll
        for (int ks = 0; ks < kH / 32; ++ks)
            acc = WMMA_BF16(lds_a_frag(&hns[col * HS_LD], ks * 32, lane),
                            glb_b_frag(Bp, ks * 32), acc);
        const float bias = out_b[n];
#pragma unroll
        for (int v = 0; v < 8; ++v) {
            const int r = v + rOff;
            const bf16 vb = (bf16)(acc[v] + bias);
            os[r * OS_LD + n] = vb;
            x_bf[(size_t)(rowBase + r) * kE + n] = vb;   // next step's x
        }
    }
    __syncthreads();

    // ---- Phase D: d_out[:, 140+t, :] = out @ regW^T + reg_b, N = 64 ---------
    if (wave < 4) {
        const int nb = wave;
        const int n  = nb * 16 + col;
        const bf16* Bp = regWb + (size_t)n * kE + kOff;
        v8f acc = {};
#pragma unroll
        for (int ks = 0; ks < kE / 32; ++ks)
            acc = WMMA_BF16(lds_a_frag(&os[col * OS_LD], ks * 32, lane),
                            glb_b_frag(Bp, ks * 32), acc);
        const float bias = reg_b[n];
#pragma unroll
        for (int v = 0; v < 8; ++v) {
            const int r = v + rOff;
            out_full[(size_t)(rowBase + r) * (kTfull * kO) + (size_t)(kTenc + t) * kO + n] =
                acc[v] + bias;
        }
    }
}

// ---------------------------------------------------------------------------
// Host orchestration (graph-capturable: only kernel launches on `stream`)
// ---------------------------------------------------------------------------
extern "C" void kernel_launch(void* const* d_in, const int* in_sizes, int n_in,
                              void* d_out, int out_size, void* d_ws, size_t ws_size,
                              hipStream_t stream) {
    const float* enc   = (const float*)d_in[0];
    const float* h0    = (const float*)d_in[1];
    const float* embW  = (const float*)d_in[2];
    const float* emb_b = (const float*)d_in[3];
    const float* w_ih  = (const float*)d_in[4];
    const float* w_hh  = (const float*)d_in[5];
    const float* b_ih  = (const float*)d_in[6];
    const float* b_hh  = (const float*)d_in[7];
    const float* outW  = (const float*)d_in[8];
    const float* out_b = (const float*)d_in[9];
    const float* regW  = (const float*)d_in[10];
    const float* reg_b = (const float*)d_in[11];
    float* out = (float*)d_out;

    // Workspace carve-up (all sizes 256B-aligned by construction, ~5.3 MB total)
    char* ws = (char*)d_ws;
    bf16*  embWb = (bf16*)ws;  ws += (size_t)kH * kE * 2;          // 512x256
    bf16*  wihb  = (bf16*)ws;  ws += (size_t)3 * kH * kH * 2;      // 1536x512
    bf16*  whhb  = (bf16*)ws;  ws += (size_t)3 * kH * kH * 2;      // 1536x512
    bf16*  outWb = (bf16*)ws;  ws += (size_t)kE * kH * 2;          // 256x512
    bf16*  regWb = (bf16*)ws;  ws += (size_t)kO * kE * 2;          // 64x256
    bf16*  x_bf  = (bf16*)ws;  ws += (size_t)kBatch * kE * 2;      // 512x256
    float* h_f32 = (float*)ws; ws += (size_t)kBatch * kH * 4;      // 512x512
    bf16*  h_bf  = (bf16*)ws;

    // 1) Convert weights to bf16 (they stay hot in the 192MB L2 afterwards)
    f32_to_bf16_kernel<<<(kH * kE + 255) / 256, 256, 0, stream>>>(embW, embWb, kH * kE);
    f32_to_bf16_kernel<<<(3 * kH * kH + 255) / 256, 256, 0, stream>>>(w_ih, wihb, 3 * kH * kH);
    f32_to_bf16_kernel<<<(3 * kH * kH + 255) / 256, 256, 0, stream>>>(w_hh, whhb, 3 * kH * kH);
    f32_to_bf16_kernel<<<(kE * kH + 255) / 256, 256, 0, stream>>>(outW, outWb, kE * kH);
    f32_to_bf16_kernel<<<(kO * kE + 255) / 256, 256, 0, stream>>>(regW, regWb, kO * kE);

    // 2) Init recurrent state (fresh every launch -> deterministic)
    init_state_kernel<<<(kBatch * kH + 255) / 256, 256, 0, stream>>>(enc, h0, x_bf, h_f32, h_bf);

    // 3) Encoder half of the regression (independent of the scan)
    encoder_reg_kernel<<<(kBatch * kTenc) / (16 * 8), 256, 0, stream>>>(enc, regWb, reg_b, out);

    // 4) Sequential GRU scan: one fused kernel per step, dec regression fused in
    for (int t = 0; t < kPred; ++t) {
        gru_step_kernel<<<kBatch / 16, 256, 0, stream>>>(
            embWb, wihb, whhb, outWb, regWb,
            emb_b, b_ih, b_hh, out_b, reg_b,
            x_bf, h_f32, h_bf, out, t);
    }
}
